// CapsuleNetwork_67508295959200
// MI455X (gfx1250) — compile-verified, matrix operand order checked by reading, not run on hardware
//
#include <hip/hip_runtime.h>
#include <hip/hip_bf16.h>
#include <math.h>

typedef _Float16 v8h  __attribute__((ext_vector_type(8)));
typedef _Float16 v16h __attribute__((ext_vector_type(16)));
typedef float    v8f  __attribute__((ext_vector_type(8)));

#define RITER 3

// Async global->LDS copy of 16 bytes (CDNA5 GLOBAL_LOAD_ASYNC_TO_LDS_B128,
// tracked by ASYNCcnt; bypasses VGPRs).  LDS byte address = low 32 bits of the
// flat pointer (ISA 10.2: LDS aperture flat addr [31:0] is the LDS address).
static __device__ inline void async_copy_b128(void* ldsDst, const void* gSrc) {
    unsigned lds = (unsigned)(size_t)ldsDst;
    asm volatile("global_load_async_to_lds_b128 %0, %1, off"
                 :: "v"(lds), "v"((unsigned long long)(size_t)gSrc)
                 : "memory");
}
static __device__ inline void wait_async_all() {
    asm volatile("s_wait_asynccnt 0x0" ::: "memory");
}

// Load a 16-half WMMA A/B fragment for this lane from a 32-half LDS row.
// Per CDNA5 ISA (16-bit A 16x32 layout): lane half h<8 -> K = 8*hi + h,
// h>=8 -> K = 16 + 8*hi + (h-8).  Two contiguous 16B ds_load_b128s.
static __device__ inline v16h ld_frag16(const _Float16* rowp, int hi) {
    v8h lo = *(const v8h*)(rowp + 8 * hi);
    v8h hh = *(const v8h*)(rowp + 16 + 8 * hi);
    v16h r;
#pragma unroll
    for (int t = 0; t < 8; ++t) { r[t] = lo[t]; r[t + 8] = hh[t]; }
    return r;
}

__global__ void k_f32_to_f16(const float* __restrict__ in, _Float16* __restrict__ out, int n) {
    int i = blockIdx.x * blockDim.x + threadIdx.x;
    if (i < n) out[i] = (_Float16)in[i];
}

// ---------------------------------------------------------------------------
// Implicit-GEMM convolution (NHWC in, HWIO weight flattened to [K,N], NHWC out)
// Block: 256 threads = 8 waves (4 M-waves x 2 N-waves).
// Block tile: 128 rows x 64 cols; each wave computes 32x32 via 2x2 WMMA subtiles.
// A tile staged with async-to-LDS copies when the 16-half segment is contiguous.
// ---------------------------------------------------------------------------
__global__ void k_conv_wmma(const _Float16* __restrict__ in, const _Float16* __restrict__ wt,
                            const float* __restrict__ bias, _Float16* __restrict__ out,
                            int IH, int IW, int Cin, int cinShift, int KH, int KW, int Cout,
                            int OH, int OW, int stride, int relu) {
    __shared__ _Float16 sA[128 * 32];   // A tile  [128 rows][32 K]
    __shared__ _Float16 sBt[64 * 32];   // B tile transposed [64 N][32 K]

    const int K    = KH * KW * Cin;
    const int Kpad = (K + 31) & ~31;
    const int tid  = threadIdx.x;
    const int lane = tid & 31;
    const int hi   = lane >> 4;
    const int ml   = lane & 15;
    const int wv   = tid >> 5;
    const int wm   = (wv & 3) * 32;     // wave row offset in tile
    const int wn   = (wv >> 2) * 32;    // wave col offset in tile
    const int rowbase = blockIdx.x * 128;
    const int nbase   = blockIdx.y * 64;

    v8f cc[2][2];
#pragma unroll
    for (int si = 0; si < 2; ++si)
#pragma unroll
        for (int sj = 0; sj < 2; ++sj)
#pragma unroll
            for (int t = 0; t < 8; ++t) cc[si][sj][t] = 0.f;

    // A staging: this thread fills one 16-half segment of one row
    const int arow = tid >> 1;
    const int akk0 = (tid & 1) * 16;
    const int grow = rowbase + arow;
    const int ow   = grow % OW;
    int tmp        = grow / OW;
    const int oh   = tmp % OH;
    const int nb   = tmp / OH;
    const _Float16* inb = in + (size_t)nb * IH * IW * Cin;

    // B staging: this thread fills 8 halfs of one K-row (transposed on store)
    const int bkk  = tid >> 3;
    const int bnn0 = (tid & 7) * 8;

    for (int kb = 0; kb < Kpad; kb += 32) {
        // ---- stage A (im2col) ----
        {
            int k0 = kb + akk0;
            int r0 = k0 >> cinShift;
            if (k0 + 15 < K && r0 == ((k0 + 15) >> cinShift)) {
                // fast path: 16 contiguous channels inside one (kh,kw) tap ->
                // two async b128 copies straight into LDS (no VGPR round trip)
                int ci  = k0 - (r0 << cinShift);
                int kw_ = r0 % KW;
                int kh_ = r0 / KW;
                const _Float16* p =
                    inb + ((size_t)(oh * stride + kh_) * IW + (ow * stride + kw_)) * Cin + ci;
                async_copy_b128(&sA[arow * 32 + akk0], p);
                async_copy_b128(&sA[arow * 32 + akk0 + 8], p + 8);
            } else {
#pragma unroll 4
                for (int e = 0; e < 16; ++e) {
                    int k = k0 + e;
                    _Float16 v = (_Float16)0.f;
                    if (k < K) {
                        int ci  = k % Cin;
                        int r   = k / Cin;
                        int kw_ = r % KW;
                        int kh_ = r / KW;
                        v = inb[((size_t)(oh * stride + kh_) * IW + (ow * stride + kw_)) * Cin + ci];
                    }
                    sA[arow * 32 + akk0 + e] = v;
                }
            }
        }
        // ---- stage B transposed ----
        {
            int k = kb + bkk;
            if (k < K) {
                v8h wvv = *(const v8h*)&wt[(size_t)k * Cout + nbase + bnn0];
#pragma unroll
                for (int e = 0; e < 8; ++e) sBt[(bnn0 + e) * 32 + bkk] = wvv[e];
            } else {
#pragma unroll
                for (int e = 0; e < 8; ++e) sBt[(bnn0 + e) * 32 + bkk] = (_Float16)0.f;
            }
            if (kb + 32 < K)
                __builtin_prefetch(&wt[(size_t)(kb + 32 + bkk) * Cout + nbase + bnn0], 0, 1);
        }
        wait_async_all();        // drain this wave's ASYNCcnt before publishing
        __syncthreads();

        v16h a0 = ld_frag16(&sA[(wm + ml) * 32], hi);
        v16h a1 = ld_frag16(&sA[(wm + 16 + ml) * 32], hi);
        v16h b0 = ld_frag16(&sBt[(wn + ml) * 32], hi);
        v16h b1 = ld_frag16(&sBt[(wn + 16 + ml) * 32], hi);
        cc[0][0] = __builtin_amdgcn_wmma_f32_16x16x32_f16(false, a0, false, b0, (short)0, cc[0][0], false, false);
        cc[0][1] = __builtin_amdgcn_wmma_f32_16x16x32_f16(false, a0, false, b1, (short)0, cc[0][1], false, false);
        cc[1][0] = __builtin_amdgcn_wmma_f32_16x16x32_f16(false, a1, false, b0, (short)0, cc[1][0], false, false);
        cc[1][1] = __builtin_amdgcn_wmma_f32_16x16x32_f16(false, a1, false, b1, (short)0, cc[1][1], false, false);
        __syncthreads();
    }

    // ---- epilogue: bias + relu, f16 store (NHWC == row-major [M,Cout]) ----
#pragma unroll
    for (int si = 0; si < 2; ++si)
#pragma unroll
        for (int sj = 0; sj < 2; ++sj) {
            int n = nbase + wn + sj * 16 + ml;
            float bv = bias ? bias[n] : 0.f;
#pragma unroll
            for (int r = 0; r < 8; ++r) {
                int row = rowbase + wm + si * 16 + r + 8 * hi;
                float x = cc[si][sj][r] + bv;
                if (relu) x = x > 0.f ? x : 0.f;
                out[(size_t)row * Cout + n] = (_Float16)x;
            }
        }
}

// ---------------------------------------------------------------------------
// Dense GEMM: out = act(A[M,K] @ B[K,N] + bias).  Same 128x64 WMMA tiling.
// Requires K % 32 == 0, N % 64 == 0.  act: 0 none, 1 relu, 2 sigmoid.
// A tile staged with async-to-LDS copies.
// ---------------------------------------------------------------------------
__global__ void k_gemm_wmma(const _Float16* __restrict__ A, const _Float16* __restrict__ Bm,
                            const float* __restrict__ bias, _Float16* __restrict__ outH,
                            float* __restrict__ outF, int M, int N, int K, int act) {
    __shared__ _Float16 sA[128 * 32];
    __shared__ _Float16 sBt[64 * 32];

    const int tid  = threadIdx.x;
    const int lane = tid & 31;
    const int hi   = lane >> 4;
    const int ml   = lane & 15;
    const int wv   = tid >> 5;
    const int wm   = (wv & 3) * 32;
    const int wn   = (wv >> 2) * 32;
    const int rowbase = blockIdx.x * 128;
    const int nbase   = blockIdx.y * 64;

    v8f cc[2][2];
#pragma unroll
    for (int si = 0; si < 2; ++si)
#pragma unroll
        for (int sj = 0; sj < 2; ++sj)
#pragma unroll
            for (int t = 0; t < 8; ++t) cc[si][sj][t] = 0.f;

    const int arow = tid >> 1;
    const int akk0 = (tid & 1) * 16;
    const int grow = rowbase + arow;
    const int bkk  = tid >> 3;
    const int bnn0 = (tid & 7) * 8;

    for (int kb = 0; kb < K; kb += 32) {
        if (grow < M) {
            const _Float16* p = &A[(size_t)grow * K + kb + akk0];
            async_copy_b128(&sA[arow * 32 + akk0], p);
            async_copy_b128(&sA[arow * 32 + akk0 + 8], p + 8);
        } else {
#pragma unroll
            for (int e = 0; e < 16; ++e) sA[arow * 32 + akk0 + e] = (_Float16)0.f;
        }
        {
            int k = kb + bkk;
            v8h wvv = *(const v8h*)&Bm[(size_t)k * N + nbase + bnn0];
#pragma unroll
            for (int e = 0; e < 8; ++e) sBt[(bnn0 + e) * 32 + bkk] = wvv[e];
            if (kb + 32 < K)
                __builtin_prefetch(&Bm[(size_t)(kb + 32 + bkk) * N + nbase + bnn0], 0, 1);
        }
        wait_async_all();
        __syncthreads();

        v16h a0 = ld_frag16(&sA[(wm + ml) * 32], hi);
        v16h a1 = ld_frag16(&sA[(wm + 16 + ml) * 32], hi);
        v16h b0 = ld_frag16(&sBt[(wn + ml) * 32], hi);
        v16h b1 = ld_frag16(&sBt[(wn + 16 + ml) * 32], hi);
        cc[0][0] = __builtin_amdgcn_wmma_f32_16x16x32_f16(false, a0, false, b0, (short)0, cc[0][0], false, false);
        cc[0][1] = __builtin_amdgcn_wmma_f32_16x16x32_f16(false, a0, false, b1, (short)0, cc[0][1], false, false);
        cc[1][0] = __builtin_amdgcn_wmma_f32_16x16x32_f16(false, a1, false, b0, (short)0, cc[1][0], false, false);
        cc[1][1] = __builtin_amdgcn_wmma_f32_16x16x32_f16(false, a1, false, b1, (short)0, cc[1][1], false, false);
        __syncthreads();
    }

#pragma unroll
    for (int si = 0; si < 2; ++si)
#pragma unroll
        for (int sj = 0; sj < 2; ++sj) {
            int n = nbase + wn + sj * 16 + ml;
            float bv = bias ? bias[n] : 0.f;
#pragma unroll
            for (int r = 0; r < 8; ++r) {
                int row = rowbase + wm + si * 16 + r + 8 * hi;
                if (row >= M) continue;
                float x = cc[si][sj][r] + bv;
                if (act == 1) x = x > 0.f ? x : 0.f;
                else if (act == 2) x = 1.f / (1.f + __expf(-x));
                if (outH) outH[(size_t)row * N + n] = (_Float16)x;
                if (outF) outF[(size_t)row * N + n] = x;
            }
        }
}

// ---------------------------------------------------------------------------
// u_hat[b,i,j,k] = sum_v w[i,j,k,v] * u[b,i,v]   (K=8 zero-padded to 32)
// grid: (mt=16, j=10, i=1152), block 32 (one wave, one WMMA).
// ---------------------------------------------------------------------------
__global__ void k_uhat_wmma(const _Float16* __restrict__ u, const _Float16* __restrict__ wc,
                            float* __restrict__ uhat) {
    const int mt = blockIdx.x, j = blockIdx.y, i = blockIdx.z;
    const int lane = threadIdx.x & 31;
    const int hi = lane >> 4, ml = lane & 15;

    v16h a, b;
#pragma unroll
    for (int t = 0; t < 16; ++t) { a[t] = (_Float16)0.f; b[t] = (_Float16)0.f; }

    if (hi == 0) {
        // halves 0..7 of hi==0 lanes cover K = 0..7 (= the only real K values)
        v8h av = *(const v8h*)&u[((size_t)(mt * 16 + ml)) * 9216 + (size_t)i * 8];
        v8h bv = *(const v8h*)&wc[(((size_t)i * 10 + j) * 16 + ml) * 8];
#pragma unroll
        for (int t = 0; t < 8; ++t) { a[t] = av[t]; b[t] = bv[t]; }
    }
    v8f c;
#pragma unroll
    for (int t = 0; t < 8; ++t) c[t] = 0.f;
    c = __builtin_amdgcn_wmma_f32_16x16x32_f16(false, a, false, b, (short)0, c, false, false);

#pragma unroll
    for (int r = 0; r < 8; ++r) {
        int brow = mt * 16 + r + 8 * hi;
        uhat[(((size_t)brow * 1152 + i) * 10 + j) * 16 + ml] = c[r];
    }
}

// ---------------------------------------------------------------------------
// Dynamic routing (R=3) + label masking.  One block per batch element.
// b_logit is recomputed incrementally from stored per-iteration v's.
// ---------------------------------------------------------------------------
__global__ void k_routing(const float* __restrict__ uhat, const int* __restrict__ y,
                          float* __restrict__ vout, _Float16* __restrict__ vflat) {
    __shared__ float s_s[160];
    __shared__ float s_v[RITER * 160];
    const int b = blockIdx.x;
    const int tid = threadIdx.x;
    const float* ub = uhat + (size_t)b * 1152 * 160;

    for (int it = 0; it < RITER; ++it) {
        if (tid < 160) s_s[tid] = 0.f;
        __syncthreads();
        for (int i = tid; i < 1152; i += 256) {
            const float* ui = ub + (size_t)i * 160;
            float lg[10];
#pragma unroll
            for (int j = 0; j < 10; ++j) {
                float l = 0.f;
                for (int t = 0; t < it; ++t) {
                    const float* vt = &s_v[t * 160 + j * 16];
                    const float* uj = ui + j * 16;
#pragma unroll
                    for (int k = 0; k < 16; ++k) l += uj[k] * vt[k];
                }
                lg[j] = l;
            }
            float mx = lg[0];
#pragma unroll
            for (int j = 1; j < 10; ++j) mx = fmaxf(mx, lg[j]);
            float sum = 0.f;
#pragma unroll
            for (int j = 0; j < 10; ++j) { lg[j] = __expf(lg[j] - mx); sum += lg[j]; }
            float inv = 1.f / sum;
#pragma unroll
            for (int j = 0; j < 10; ++j) {
                float cj = lg[j] * inv;
                const float* uj = ui + j * 16;
#pragma unroll
                for (int k = 0; k < 16; ++k) atomicAdd(&s_s[j * 16 + k], cj * uj[k]);
            }
        }
        __syncthreads();
        if (tid < 10) {
            int j = tid;
            float n2 = 0.f;
#pragma unroll
            for (int k = 0; k < 16; ++k) { float x = s_s[j * 16 + k]; n2 += x * x; }
            float n = sqrtf(n2);
            float f = n2 / ((1.f + n2) * (n + 1e-7f));
#pragma unroll
            for (int k = 0; k < 16; ++k) s_v[it * 160 + j * 16 + k] = f * s_s[j * 16 + k];
        }
        __syncthreads();
    }
    if (tid < 160) {
        float v = s_v[(RITER - 1) * 160 + tid];
        vout[(size_t)b * 160 + tid] = v;                    // v output [B,1,10,16]
        int j = tid >> 4;
        float m = (j == y[b]) ? v : 0.f;                    // mask by true label
        vflat[(size_t)b * 160 + tid] = (_Float16)m;
    }
}

// ---------------------------------------------------------------------------
extern "C" void kernel_launch(void* const* d_in, const int* in_sizes, int n_in,
                              void* d_out, int out_size, void* d_ws, size_t ws_size,
                              hipStream_t stream) {
    const float* x   = (const float*)d_in[0];
    const int*   y   = (const int*)d_in[1];
    const float* c1k = (const float*)d_in[2];  const float* c1b = (const float*)d_in[3];
    const float* c2k = (const float*)d_in[4];  const float* c2b = (const float*)d_in[5];
    const float* c3k = (const float*)d_in[6];  const float* c3b = (const float*)d_in[7];
    const float* pk  = (const float*)d_in[8];  const float* pb  = (const float*)d_in[9];
    const float* wcp = (const float*)d_in[10];
    const float* d1w = (const float*)d_in[11]; const float* d1b = (const float*)d_in[12];
    const float* d2w = (const float*)d_in[13]; const float* d2b = (const float*)d_in[14];
    const float* d3w = (const float*)d_in[15]; const float* d3b = (const float*)d_in[16];
    float* out = (float*)d_out;

    const int B = 256;
    char* ws = (char*)d_ws;
    size_t off = 0;
    auto alloc = [&](size_t bytes) -> void* {
        off = (off + 255) & ~(size_t)255;
        void* p = ws + off;
        off += bytes;
        return p;
    };

    // f16 buffers
    _Float16* xh  = (_Float16*)alloc((size_t)B * 1600 * 2);
    _Float16* w1h = (_Float16*)alloc((size_t)1600 * 2);
    _Float16* w2h = (_Float16*)alloc((size_t)663552 * 2);
    _Float16* w3h = (_Float16*)alloc((size_t)2654208 * 2);
    _Float16* wph = (_Float16*)alloc((size_t)5308416 * 2);
    _Float16* wch = (_Float16*)alloc((size_t)1474560 * 2);
    _Float16* wd1 = (_Float16*)alloc((size_t)163840 * 2);
    _Float16* wd2 = (_Float16*)alloc((size_t)2097152 * 2);
    _Float16* wd3 = (_Float16*)alloc((size_t)3276800 * 2);
    _Float16* a1  = (_Float16*)alloc((size_t)B * 36 * 36 * 64 * 2);
    _Float16* a2  = (_Float16*)alloc((size_t)B * 28 * 28 * 128 * 2);
    _Float16* a3  = (_Float16*)alloc((size_t)B * 20 * 20 * 256 * 2);
    _Float16* ap  = (_Float16*)alloc((size_t)B * 9216 * 2);           // prim caps == u
    _Float16* vfl = (_Float16*)alloc((size_t)B * 160 * 2);
    _Float16* r1  = (_Float16*)alloc((size_t)B * 1024 * 2);
    _Float16* r2  = (_Float16*)alloc((size_t)B * 2048 * 2);
    float*    uhat = (float*)alloc((size_t)B * 1152 * 160 * 4);

    auto cvt = [&](const float* s, _Float16* d, int n) {
        k_f32_to_f16<<<(n + 255) / 256, 256, 0, stream>>>(s, d, n);
    };
    cvt(x, xh, B * 1600);
    cvt(c1k, w1h, 1600);
    cvt(c2k, w2h, 663552);
    cvt(c3k, w3h, 2654208);
    cvt(pk, wph, 5308416);
    cvt(wcp, wch, 1474560);
    cvt(d1w, wd1, 163840);
    cvt(d2w, wd2, 2097152);
    cvt(d3w, wd3, 3276800);

    // conv stack (implicit GEMM, f16 WMMA, f32 accumulate)
    k_conv_wmma<<<dim3((B * 36 * 36) / 128, 64 / 64), 256, 0, stream>>>(
        xh, w1h, c1b, a1, 40, 40, 1, 0, 5, 5, 64, 36, 36, 1, 1);
    k_conv_wmma<<<dim3((B * 28 * 28) / 128, 128 / 64), 256, 0, stream>>>(
        a1, w2h, c2b, a2, 36, 36, 64, 6, 9, 9, 128, 28, 28, 1, 1);
    k_conv_wmma<<<dim3((B * 20 * 20) / 128, 256 / 64), 256, 0, stream>>>(
        a2, w3h, c3b, a3, 28, 28, 128, 7, 9, 9, 256, 20, 20, 1, 1);
    k_conv_wmma<<<dim3((B * 6 * 6) / 128, 256 / 64), 256, 0, stream>>>(
        a3, wph, pb, ap, 20, 20, 256, 8, 9, 9, 256, 6, 6, 2, 0);   // linear

    // pose predictions
    k_uhat_wmma<<<dim3(16, 10, 1152), 32, 0, stream>>>(ap, wch, uhat);

    // dynamic routing + mask; writes v (first 40960 floats of d_out) and masked v_flat (f16)
    k_routing<<<B, 256, 0, stream>>>(uhat, y, out, vfl);

    // reconstruction decoder
    k_gemm_wmma<<<dim3(2, 1024 / 64), 256, 0, stream>>>(vfl, wd1, d1b, r1, nullptr,
                                                        B, 1024, 160, 1);
    k_gemm_wmma<<<dim3(2, 2048 / 64), 256, 0, stream>>>(r1, wd2, d2b, r2, nullptr,
                                                        B, 2048, 1024, 1);
    k_gemm_wmma<<<dim3(2, 1600 / 64), 256, 0, stream>>>(r2, wd3, d3b, nullptr,
                                                        out + (size_t)B * 160,
                                                        B, 1600, 2048, 2);   // sigmoid -> recon
}
